// LiquidCell_25795573580554
// MI455X (gfx1250) — compile-verified, matrix operand order checked by reading, not run on hardware
//
#include <hip/hip_runtime.h>
#include <hip/hip_bf16.h>
#include <math.h>

// ---------------------------------------------------------------------------
// LiquidCell fused ODE kernel for gfx1250 (MI455X).
//  - whole 9-step Euler integration fused in one kernel, h resident in LDS
//  - GEMMs via v_wmma_f32_16x16x32_bf16, 16-row M-tile per wave32
//  - weights converted once to bf16 in d_ws, stream from L2 every step
//  - LN stats accumulated in registers from WMMA accumulators (shfl butterfly)
//  - bf16 conversions via native v_cvt_pk_bf16_f32 when available
// ---------------------------------------------------------------------------

typedef __attribute__((ext_vector_type(16))) __bf16 v16bf;
typedef __attribute__((ext_vector_type(2)))  __bf16 v2bf;
typedef __attribute__((ext_vector_type(8)))  float  v8f;

#define HD __device__ __forceinline__

#define B_TOT 131072
#define H_    256
#define IN_   256
#define K1_   512     // H + IN
#define N1_   1024    // 4H
#define N2_   512     // 2H
#define N3_   256     // H
#define ROWS_ 64      // rows per workgroup (4 waves x 16)
#define BDIM_ 128

// ---- bf16 conversion helpers (prefer native packed cvt) --------------------
HD unsigned pk_bf16(float lo, float hi) {
#if defined(__has_builtin) && __has_builtin(__builtin_amdgcn_cvt_pk_bf16_f32)
  union { v2bf v; unsigned u; } c;
  c.v = __builtin_amdgcn_cvt_pk_bf16_f32(lo, hi);
  return c.u;
#else
  union { __bf16 b[2]; unsigned u; } c;
  c.b[0] = (__bf16)lo; c.b[1] = (__bf16)hi;
  return c.u;
#endif
}
HD unsigned short f2bf(float f) {
  union { __bf16 b; unsigned short u; } c;
  c.b = (__bf16)f;
  return c.u;
}
HD float bf2f(unsigned short h) { return __uint_as_float(((unsigned)h) << 16); }

HD float gelu_exact(float x) { return 0.5f * x * (1.0f + erff(x * 0.7071067811865475f)); }

// 16-bit A/B fragment: 16 bf16 per lane as two contiguous 16B chunks
HD v16bf ld_frag(const unsigned short* p) {
  union { uint4 q[2]; v16bf v; } f;
  f.q[0] = *(const uint4*)(p);
  f.q[1] = *(const uint4*)(p + 16);
  return f.v;
}

// Accumulate a 16x64 output strip (4 n-tiles) over `KB` k-blocks of 32.
// arow  : A row base for this lane (LDS, row already selected, element units)
// wbase : weight base + k-offset (global bf16, [N][K] row-major)
template <int KB>
HD void gemm_part(const unsigned short* __restrict__ arow,
                  const unsigned short* __restrict__ wbase, size_t wstride,
                  int n0, int nl, int kofs, v8f* acc) {
  for (int kb = 0; kb < KB; ++kb) {
    v16bf a = ld_frag(arow + kb * 32 + kofs);
    __builtin_prefetch((const void*)(wbase + (size_t)(n0 + nl) * wstride + kb * 32 + 512), 0, 0);
#pragma unroll
    for (int t4 = 0; t4 < 4; ++t4) {
      const int n = n0 + t4 * 16 + nl;
      v16bf b = ld_frag(wbase + (size_t)n * wstride + kb * 32 + kofs);
      acc[t4] = __builtin_amdgcn_wmma_f32_16x16x32_bf16(
          false, a, false, b, (short)0, acc[t4], false, false);
    }
  }
}

// ---------------------------------------------------------------------------
// f32 -> packed bf16 conversion (pairs)
// ---------------------------------------------------------------------------
__global__ void cvt_f32_bf16(const float* __restrict__ in,
                             unsigned short* __restrict__ outp, int npairs) {
  int i = blockIdx.x * blockDim.x + threadIdx.x;
  if (i < npairs) {
    float2 v = ((const float2*)in)[i];
    ((unsigned*)outp)[i] = pk_bf16(v.x, v.y);
  }
}

// ---------------------------------------------------------------------------
// Main fused kernel
// ---------------------------------------------------------------------------
__launch_bounds__(BDIM_, 1)
__global__ void liquid_kernel(const unsigned short* __restrict__ xbf,
                              const float* __restrict__ hin,
                              const unsigned short* __restrict__ W1b,
                              const unsigned short* __restrict__ W2b,
                              const unsigned short* __restrict__ W3b,
                              const float* __restrict__ b1, const float* __restrict__ g1,
                              const float* __restrict__ be1,
                              const float* __restrict__ b2, const float* __restrict__ b3,
                              const float* __restrict__ wt, const float* __restrict__ bt,
                              const float* __restrict__ go, const float* __restrict__ bo,
                              float* __restrict__ out) {
  __shared__ float p_b1[N1_], p_g1[N1_], p_be1[N1_];
  __shared__ float p_b2[N2_], p_b3[N3_];
  __shared__ float p_wt[H_], p_bt[H_], p_go[H_], p_bo[H_];
  __shared__ float stat_s[BDIM_ / 32][16][2];  // per-wave, per-row {sum, sumsq}
  __shared__ unsigned short xs[ROWS_ * IN_];   // x tile, bf16
  __shared__ unsigned short hs[ROWS_ * H_];    // h state, bf16
  __shared__ unsigned short z1s[ROWS_ * N1_];  // 4H activation
  __shared__ unsigned short z2s[ROWS_ * N2_];  // 2H activation

  const int tid  = threadIdx.x;
  const int lane = tid & 31;
  const int wv   = tid >> 5;
  const int nl   = lane & 15;   // column-in-tile / A row lane
  const int lg   = lane >> 4;   // lane half
  const int m0   = wv * 16;     // wave's local row base
  const int rowA = m0 + nl;     // this lane's A row
  const int kofs = lg * 8;      // K offset of lane half within fragment
  const size_t rowg0 = (size_t)blockIdx.x * ROWS_;

  // ---- preload params + tiles ----
  for (int i = tid; i < N1_; i += BDIM_) { p_b1[i] = b1[i]; p_g1[i] = g1[i]; p_be1[i] = be1[i]; }
  for (int i = tid; i < N2_; i += BDIM_) { p_b2[i] = b2[i]; }
  for (int i = tid; i < N3_; i += BDIM_) {
    p_b3[i] = b3[i]; p_wt[i] = wt[i]; p_bt[i] = bt[i]; p_go[i] = go[i]; p_bo[i] = bo[i];
  }
  {
    const uint4* gx = (const uint4*)(xbf + rowg0 * IN_);
    uint4* lx = (uint4*)xs;
    for (int i = tid; i < (ROWS_ * IN_) / 8; i += BDIM_) lx[i] = gx[i];
  }
  {
    const float2* gh = (const float2*)(hin + rowg0 * H_);
    unsigned* lh = (unsigned*)hs;
    for (int i = tid; i < (ROWS_ * H_) / 2; i += BDIM_) {
      float2 v = gh[i];
      lh[i] = pk_bf16(v.x, v.y);
    }
  }
  __syncthreads();

  const float dtv = 1.0f / 9.0f;

  for (int s = 0; s < 9; ++s) {
    const float tcur = (float)s * dtv;

    // ---------------- GEMM1: [16,512] x [512,1024] + b1 -> z1 (pre-LN) -----
    // LN statistics accumulated in registers straight from the accumulators:
    // lane (nl,lg) sees rows {lg*8+v} at columns n === nl (mod 16).
    float rsum[8], rsq[8];
#pragma unroll
    for (int v = 0; v < 8; ++v) { rsum[v] = 0.f; rsq[v] = 0.f; }

    for (int nc = 0; nc < N1_ / 64; ++nc) {
      const int n0 = nc * 64;
      v8f acc[4];
#pragma unroll
      for (int t4 = 0; t4 < 4; ++t4) {
        const float bv = p_b1[n0 + t4 * 16 + nl];
#pragma unroll
        for (int v = 0; v < 8; ++v) acc[t4][v] = bv;
      }
      gemm_part<8>(hs + (size_t)rowA * H_,  W1b,       K1_, n0, nl, kofs, acc); // K 0..255 (h)
      gemm_part<8>(xs + (size_t)rowA * IN_, W1b + 256, K1_, n0, nl, kofs, acc); // K 256..511 (x)
#pragma unroll
      for (int t4 = 0; t4 < 4; ++t4) {
        const int n = n0 + t4 * 16 + nl;
#pragma unroll
        for (int v = 0; v < 8; ++v) {
          const float val = acc[t4][v];
          rsum[v] += val;
          rsq[v]  += val * val;
          z1s[(size_t)(m0 + v + 8 * lg) * N1_ + n] = f2bf(val);
        }
      }
    }
    // butterfly-reduce stats over the 16 n-lanes; publish via tiny LDS scratch
#pragma unroll
    for (int v = 0; v < 8; ++v) {
      float s0 = rsum[v], q0 = rsq[v];
      s0 += __shfl_xor(s0, 1, 32);  q0 += __shfl_xor(q0, 1, 32);
      s0 += __shfl_xor(s0, 2, 32);  q0 += __shfl_xor(q0, 2, 32);
      s0 += __shfl_xor(s0, 4, 32);  q0 += __shfl_xor(q0, 4, 32);
      s0 += __shfl_xor(s0, 8, 32);  q0 += __shfl_xor(q0, 8, 32);
      if (nl == 0) { stat_s[wv][lg * 8 + v][0] = s0; stat_s[wv][lg * 8 + v][1] = q0; }
    }
    __syncthreads();

    // ---------------- LayerNorm(4H) + GELU on z1 (single pass) -------------
    {
      const int row = m0 + nl;                    // 2 lanes per row (lg halves)
      const float mu   = stat_s[wv][nl][0] * (1.0f / (float)N1_);
      const float var  = stat_s[wv][nl][1] * (1.0f / (float)N1_) - mu * mu;
      const float rstd = rsqrtf(var + 1e-5f);
      unsigned* zr = (unsigned*)(z1s + (size_t)row * N1_ + lg * 512);
      const float* gg = p_g1  + lg * 512;
      const float* bb = p_be1 + lg * 512;
      for (int j = 0; j < 256; ++j) {
        unsigned u = zr[j];
        float a0 = (bf2f((unsigned short)u)       - mu) * rstd * gg[2*j]   + bb[2*j];
        float a1 = (bf2f((unsigned short)(u>>16)) - mu) * rstd * gg[2*j+1] + bb[2*j+1];
        a0 = gelu_exact(a0); a1 = gelu_exact(a1);
        zr[j] = pk_bf16(a0, a1);
      }
    }
    __syncthreads();

    // ---------------- GEMM2: [16,1024] x [1024,512] + b2, GELU -> z2 -------
    for (int nc = 0; nc < N2_ / 64; ++nc) {
      const int n0 = nc * 64;
      v8f acc[4];
#pragma unroll
      for (int t4 = 0; t4 < 4; ++t4) {
        const float bv = p_b2[n0 + t4 * 16 + nl];
#pragma unroll
        for (int v = 0; v < 8; ++v) acc[t4][v] = bv;
      }
      gemm_part<32>(z1s + (size_t)rowA * N1_, W2b, N1_, n0, nl, kofs, acc);
#pragma unroll
      for (int t4 = 0; t4 < 4; ++t4) {
        const int n = n0 + t4 * 16 + nl;
#pragma unroll
        for (int v = 0; v < 8; ++v)
          z2s[(size_t)(m0 + v + 8 * lg) * N2_ + n] = f2bf(gelu_exact(acc[t4][v]));
      }
    }
    __syncthreads();

    // ---------------- GEMM3: [16,512] x [512,256] + b3; h update -----------
    for (int nc = 0; nc < N3_ / 64; ++nc) {
      const int n0 = nc * 64;
      v8f acc[4];
#pragma unroll
      for (int t4 = 0; t4 < 4; ++t4) {
        const float bv = p_b3[n0 + t4 * 16 + nl];
#pragma unroll
        for (int v = 0; v < 8; ++v) acc[t4][v] = bv;
      }
      gemm_part<16>(z2s + (size_t)rowA * N2_, W3b, N2_, n0, nl, kofs, acc);
#pragma unroll
      for (int t4 = 0; t4 < 4; ++t4) {
        const int n = n0 + t4 * 16 + nl;
        const float tmv = tanhf(tcur * p_wt[n] + p_bt[n]);
#pragma unroll
        for (int v = 0; v < 8; ++v) {
          const size_t idx = (size_t)(m0 + v + 8 * lg) * H_ + n;
          float hv = bf2f(hs[idx]) + dtv * acc[t4][v] * tmv;
          hs[idx] = f2bf(hv);
        }
      }
    }
    __syncthreads();
  }

  // ---------------- final LayerNorm(H) -> out ------------------------------
  {
    const int row = m0 + nl;
    const unsigned short* hr = hs + (size_t)row * H_ + lg * 128;
    const unsigned* hu = (const unsigned*)hr;
    float sum = 0.f, sq = 0.f;
    for (int j = 0; j < 64; ++j) {
      unsigned u = hu[j];
      float a0 = bf2f((unsigned short)u), a1 = bf2f((unsigned short)(u >> 16));
      sum += a0 + a1; sq += a0 * a0 + a1 * a1;
    }
    sum += __shfl_xor(sum, 16, 32);
    sq  += __shfl_xor(sq, 16, 32);
    const float mu   = sum * (1.0f / (float)H_);
    const float rstd = rsqrtf(sq * (1.0f / (float)H_) - mu * mu + 1e-5f);
    float* og = out + (rowg0 + row) * H_ + lg * 128;
    const float* gg = p_go + lg * 128;
    const float* bb = p_bo + lg * 128;
    for (int j = 0; j < 128; ++j)
      og[j] = (bf2f(hr[j]) - mu) * rstd * gg[j] + bb[j];
  }
}

// ---------------------------------------------------------------------------
extern "C" void kernel_launch(void* const* d_in, const int* in_sizes, int n_in,
                              void* d_out, int out_size, void* d_ws, size_t ws_size,
                              hipStream_t stream) {
  const float* x   = (const float*)d_in[0];
  const float* h   = (const float*)d_in[1];
  const float* W1  = (const float*)d_in[2];
  const float* b1  = (const float*)d_in[3];
  const float* g1  = (const float*)d_in[4];
  const float* be1 = (const float*)d_in[5];
  const float* W2  = (const float*)d_in[6];
  const float* b2  = (const float*)d_in[7];
  const float* W3  = (const float*)d_in[8];
  const float* b3  = (const float*)d_in[9];
  const float* wt  = (const float*)d_in[10];
  const float* bt  = (const float*)d_in[11];
  const float* go  = (const float*)d_in[12];
  const float* bo  = (const float*)d_in[13];
  float* out = (float*)d_out;

  // workspace layout (bf16 elements)
  unsigned short* xbf = (unsigned short*)d_ws;                   // B*IN
  unsigned short* W1b = xbf + (size_t)B_TOT * IN_;               // N1*K1
  unsigned short* W2b = W1b + (size_t)N1_ * K1_;                 // N2*N1
  unsigned short* W3b = W2b + (size_t)N2_ * N1_;                 // N3*N2

  // f32 -> bf16 conversions (re-run every call; deterministic)
  cvt_f32_bf16<<<(B_TOT * IN_ / 2 + 255) / 256, 256, 0, stream>>>(x,  xbf, B_TOT * IN_ / 2);
  cvt_f32_bf16<<<(N1_ * K1_ / 2 + 255) / 256, 256, 0, stream>>>(W1, W1b, N1_ * K1_ / 2);
  cvt_f32_bf16<<<(N2_ * N1_ / 2 + 255) / 256, 256, 0, stream>>>(W2, W2b, N2_ * N1_ / 2);
  cvt_f32_bf16<<<(N3_ * N2_ / 2 + 255) / 256, 256, 0, stream>>>(W3, W3b, N3_ * N2_ / 2);

  liquid_kernel<<<B_TOT / ROWS_, BDIM_, 0, stream>>>(
      xbf, h, W1b, W2b, W3b, b1, g1, be1, b2, b3, wt, bt, go, bo, out);
}
